// LrUpsampling_77618648973642
// MI455X (gfx1250) — matrix operand order repaired and checked by reading, not compiled, not run on hardware
//
#include <hip/hip_runtime.h>

#define LRD   512
#define HRD   2048
#define NHEAD 4
#define CD    512        // per-head channels
#define ND    4096       // nodes
#define EPSV  1e-5f

typedef __attribute__((ext_vector_type(16))) __bf16 v16bf;
typedef __attribute__((ext_vector_type(8)))  float  v8f;

struct alignas(16) B128 { unsigned int x, y, z, w; };
union FragAB { v16bf v; B128 q[2]; unsigned short u[16]; };

__device__ __forceinline__ unsigned short f2bf(float f) {
    union { float f; unsigned u; } x; x.f = f;
    unsigned r = x.u + 0x7fffu + ((x.u >> 16) & 1u);   // round-to-nearest-even
    return (unsigned short)(r >> 16);
}

// A-matrix fragment (16x32 bf16) from row-major [M][K] source (rows 16, ld elems).
// ISA 7.12.2: lanes 0-15 row M=lane, elems {K..K+7, K+16..K+23}; lanes 16-31 +8.
__device__ __forceinline__ v16bf load_a_bf16(const unsigned short* base, int ld, int kc) {
    const int lane = threadIdx.x & 31;
    const int row  = lane & 15;
    const int kh   = (lane >> 4) << 3;
    const unsigned short* p = base + (size_t)row * ld + kc + kh;
    FragAB f;
    f.q[0] = *(const B128*)(p);
    f.q[1] = *(const B128*)(p + 16);
    return f.v;
}

// B-matrix fragment (32x16 bf16, B[K][N]) sourced from row-major [N][K] storage:
// lanes 0-15: N=lane, K={kc..kc+7, kc+16..kc+23} ; lanes 16-31: N=lane-16, K={kc+8..kc+15, kc+24..kc+31}.
__device__ __forceinline__ v16bf load_bt_bf16(const unsigned short* base, int ld, int kc) {
    const int lane = threadIdx.x & 31;
    const int row  = lane & 15;
    const int kh   = (lane >> 4) << 3;
    const unsigned short* p = base + (size_t)row * ld + kc + kh;
    FragAB f;
    f.q[0] = *(const B128*)(p);
    f.q[1] = *(const B128*)(p + 16);
    return f.v;
}

__device__ __forceinline__ v8f wmma_bf16(v16bf a, v16bf b, v8f c) {
    return __builtin_amdgcn_wmma_f32_16x16x32_bf16(false, a, false, b, (short)0, c, false, false);
}

// ---------------- pack kernels ----------------
// x_bf16[n][l] = bf16(lr_x[l][n])   (lr_x is [LR][N])
__global__ void pack_x_kernel(const float* __restrict__ lr_x, unsigned short* __restrict__ xb) {
    int idx = blockIdx.x * blockDim.x + threadIdx.x;     // over N*LR
    int n = idx / LRD, l = idx - n * LRD;
    xb[idx] = f2bf(lr_x[(size_t)l * ND + n]);
}

// Wt[j][l] = bf16(W[l][j])   (W is [LR][HR])
__global__ void pack_wt_kernel(const float* __restrict__ W, unsigned short* __restrict__ Wt) {
    int idx = blockIdx.x * blockDim.x + threadIdx.x;     // over HR*LR
    int j = idx / LRD, l = idx - j * LRD;
    Wt[idx] = f2bf(W[(size_t)l * HRD + j]);
}

// ---------------- projection GEMM:  out = X[4096x512] @ W[512x2048] + b ----------------
// 8 waves / workgroup; wave w does rows m0+16w, 64 cols; modes: 0 bf16 [n][HR],
// 1 bf16 transposed [j][N] (for V), 2 f32 [n][HR] (skip).
__global__ __launch_bounds__(256) void proj_gemm_kernel(
        const unsigned short* __restrict__ X, const unsigned short* __restrict__ Wt,
        const float* __restrict__ bias, void* __restrict__ out, int mode) {
    const int w = threadIdx.x >> 5, lane = threadIdx.x & 31;
    const int m0 = blockIdx.x * 128 + w * 16;
    const int j0 = blockIdx.y * 64;
    v8f acc[4] = {};
    const unsigned short* Xb = X + (size_t)m0 * LRD;
    for (int kc = 0; kc < LRD; kc += 32) {
        v16bf a = load_a_bf16(Xb, LRD, kc);
        for (int i = 0; i < 4; ++i) {
            v16bf b = load_bt_bf16(Wt + (size_t)(j0 + i * 16) * LRD, LRD, kc);
            acc[i] = wmma_bf16(a, b, acc[i]);
        }
    }
    for (int i = 0; i < 4; ++i) {
        const int j = j0 + i * 16 + (lane & 15);
        const float bv = bias[j];
        for (int r = 0; r < 8; ++r) {
            const int row = m0 + r + ((lane >> 4) << 3);
            const float val = acc[i][r] + bv;
            if (mode == 0)      ((unsigned short*)out)[(size_t)row * HRD + j] = f2bf(val);
            else if (mode == 1) ((unsigned short*)out)[(size_t)j * ND + row]  = f2bf(val);
            else                ((float*)out)[(size_t)row * HRD + j] = val;
        }
    }
}

// ---------------- flash attention (one head x 16 dst rows per workgroup) ----------------
__global__ __launch_bounds__(256) void attn_kernel(
        const unsigned short* __restrict__ Q, const unsigned short* __restrict__ K,
        const unsigned short* __restrict__ VT, float* __restrict__ OS) {
    const int h = blockIdx.y;
    const int n0 = blockIdx.x * 16;
    const int w = threadIdx.x >> 5, lane = threadIdx.x & 31;
    const int cbase = w * 64;

    __shared__ alignas(16) unsigned short Plds[16 * 128];
    __shared__ float pmax[8 * 16], psum[8 * 16];
    __shared__ float run_max[16], run_sum[16], corr[16], gmax[16];

    if (threadIdx.x < 16) { run_max[threadIdx.x] = -__builtin_inff(); run_sum[threadIdx.x] = 0.f; }
    __syncthreads();

    const unsigned short* Qb = Q + (size_t)n0 * HRD + h * CD;
    const float scale = 0.044194173824159216f;   // 1/sqrt(512)
    v8f acc[4] = {};

    for (int m0 = 0; m0 < ND; m0 += 128) {
        // ---- S tile: rows n0..n0+15, cols m0+16w..+15, reduce over C=512
        v8f s = {};
        const unsigned short* Kb = K + (size_t)(m0 + w * 16) * HRD + h * CD;
        for (int kc = 0; kc < CD; kc += 32) {
            v16bf a = load_a_bf16(Qb, HRD, kc);
            v16bf b = load_bt_bf16(Kb, HRD, kc);
            s = wmma_bf16(a, b, s);
        }
        // ---- per-wave row max over its 16 columns
        float sreg[8], rmax[8];
        for (int r = 0; r < 8; ++r) {
            float v = s[r] * scale;
            sreg[r] = v;
            float m = v;
            for (int off = 1; off < 16; off <<= 1) m = fmaxf(m, __shfl_xor(m, off));
            rmax[r] = m;
        }
        if (lane == 0)  for (int r = 0; r < 8; ++r) pmax[w * 16 + r]     = rmax[r];
        if (lane == 16) for (int r = 0; r < 8; ++r) pmax[w * 16 + 8 + r] = rmax[r];
        __syncthreads();
        // ---- online stats update (16 rows, one thread each)
        if (threadIdx.x < 16) {
            const int t = threadIdx.x;
            float bm = pmax[t];
            for (int ww = 1; ww < 8; ++ww) bm = fmaxf(bm, pmax[ww * 16 + t]);
            const float nm = fmaxf(run_max[t], bm);
            const float c  = __expf(run_max[t] - nm);
            run_max[t] = nm; corr[t] = c; run_sum[t] *= c; gmax[t] = nm;
        }
        __syncthreads();
        // ---- P = exp(S - max), block row sums, rescale output accumulators
        float rs[8];
        for (int r = 0; r < 8; ++r) {
            const int row = r + ((lane >> 4) << 3);
            const float p = __expf(sreg[r] - gmax[row]);
            Plds[row * 128 + w * 16 + (lane & 15)] = f2bf(p);
            float su = p;
            for (int off = 1; off < 16; off <<= 1) su += __shfl_xor(su, off);
            rs[r] = su;
            const float c = corr[row];
            for (int i = 0; i < 4; ++i) acc[i][r] = acc[i][r] * c;
        }
        if (lane == 0)  for (int r = 0; r < 8; ++r) psum[w * 16 + r]     = rs[r];
        if (lane == 16) for (int r = 0; r < 8; ++r) psum[w * 16 + 8 + r] = rs[r];
        __syncthreads();
        if (threadIdx.x < 16) {
            const int t = threadIdx.x;
            float a = 0.f;
            for (int ww = 0; ww < 8; ++ww) a += psum[ww * 16 + t];
            run_sum[t] += a;
        }
        // ---- P @ V : each wave owns 64 channels
        for (int kb = 0; kb < 4; ++kb) {
            v16bf pa = load_a_bf16(Plds, 128, kb * 32);
            for (int i = 0; i < 4; ++i) {
                const unsigned short* Vb = VT + (size_t)(h * CD + cbase + i * 16) * ND + m0;
                v16bf b = load_bt_bf16(Vb, ND, kb * 32);
                acc[i] = wmma_bf16(pa, b, acc[i]);
            }
        }
        __syncthreads();
    }
    // ---- epilogue: out = acc / rowsum + skip (OS already holds x@Wskip + b)
    for (int r = 0; r < 8; ++r) {
        const int row = r + ((lane >> 4) << 3);
        const float inv = 1.f / run_sum[row];
        for (int i = 0; i < 4; ++i) {
            const size_t idx = (size_t)(n0 + row) * HRD + h * CD + cbase + i * 16 + (lane & 15);
            OS[idx] += acc[i][r] * inv;
        }
    }
}

// ---------------- GraphNorm + unnormalized Y (bf16, [HR][N]) + per-row rnorm ----------------
__global__ void gn_kernel(const float* __restrict__ OS, const float* __restrict__ gamma,
                          const float* __restrict__ beta, const float* __restrict__ mscale,
                          unsigned short* __restrict__ Y, float* __restrict__ rnorm) {
    const int j = blockIdx.x * blockDim.x + threadIdx.x;
    if (j >= HRD) return;
    float sum = 0.f, sumsq = 0.f;
    for (int n = 0; n < ND; ++n) {
        const float v = OS[(size_t)n * HRD + j];
        sum += v; sumsq += v * v;
    }
    const float mean = sum * (1.f / ND), ex2 = sumsq * (1.f / ND);
    const float s = mscale[j];
    const float var = ex2 - 2.f * s * mean * mean + s * s * mean * mean;
    const float istd = rsqrtf(var + EPSV);
    const float g = gamma[j], b = beta[j], sm = s * mean;
    float nsq = 0.f;
    for (int n = 0; n < ND; ++n) {
        const float v = OS[(size_t)n * HRD + j];
        const float gn = (v - sm) * istd * g + b;
        nsq += gn * gn;
        Y[(size_t)j * ND + n] = f2bf(gn);
    }
    rnorm[j] = rsqrtf(nsq);
}

// ---------------- gram: G = relu( rnorm_i * rnorm_j * (Y Y^T) ), [HR][HR] f32 ----------------
__global__ __launch_bounds__(256) void gram_kernel(const unsigned short* __restrict__ Y,
                                                   const float* __restrict__ rnorm,
                                                   float* __restrict__ G) {
    const int w = threadIdx.x >> 5, lane = threadIdx.x & 31;
    const int i0 = blockIdx.x * 128 + w * 16;
    const int j0 = blockIdx.y * 64;
    v8f acc[4] = {};
    const unsigned short* Ab = Y + (size_t)i0 * ND;
    for (int kc = 0; kc < ND; kc += 32) {
        v16bf a = load_a_bf16(Ab, ND, kc);
        for (int t = 0; t < 4; ++t) {
            v16bf b = load_bt_bf16(Y + (size_t)(j0 + t * 16) * ND, ND, kc);
            acc[t] = wmma_bf16(a, b, acc[t]);
        }
    }
    for (int t = 0; t < 4; ++t) {
        const int j = j0 + t * 16 + (lane & 15);
        const float rj = rnorm[j];
        for (int r = 0; r < 8; ++r) {
            const int i = i0 + r + ((lane >> 4) << 3);
            const float v = acc[t][r] * rnorm[i] * rj;
            G[(size_t)i * HRD + j] = fmaxf(v, 0.f);
        }
    }
}

extern "C" void kernel_launch(void* const* d_in, const int* in_sizes, int n_in,
                              void* d_out, int out_size, void* d_ws, size_t ws_size,
                              hipStream_t stream) {
    (void)in_sizes; (void)n_in; (void)out_size; (void)ws_size;
    const float* lr_x   = (const float*)d_in[0];
    const float* Wq     = (const float*)d_in[1];
    const float* bq     = (const float*)d_in[2];
    const float* Wk     = (const float*)d_in[3];
    const float* bk     = (const float*)d_in[4];
    const float* Wv     = (const float*)d_in[5];
    const float* bv     = (const float*)d_in[6];
    const float* Wskip  = (const float*)d_in[7];
    const float* bskip  = (const float*)d_in[8];
    const float* gamma  = (const float*)d_in[9];
    const float* beta   = (const float*)d_in[10];
    const float* mscale = (const float*)d_in[11];
    float* G = (float*)d_out;

    char* ws = (char*)d_ws;
    size_t off = 0;
    auto carve = [&](size_t bytes) { void* p = ws + off; off = (off + bytes + 255) & ~(size_t)255; return p; };
    unsigned short* Xb  = (unsigned short*)carve((size_t)ND * LRD * 2);   //  4 MB
    unsigned short* WtQ = (unsigned short*)carve((size_t)HRD * LRD * 2);  //  2 MB
    unsigned short* WtK = (unsigned short*)carve((size_t)HRD * LRD * 2);
    unsigned short* WtV = (unsigned short*)carve((size_t)HRD * LRD * 2);
    unsigned short* WtS = (unsigned short*)carve((size_t)HRD * LRD * 2);
    unsigned short* Qb  = (unsigned short*)carve((size_t)ND * HRD * 2);   // 16 MB
    unsigned short* Kb  = (unsigned short*)carve((size_t)ND * HRD * 2);   // 16 MB
    unsigned short* VT  = (unsigned short*)carve((size_t)HRD * ND * 2);   // 16 MB
    float*          OS  = (float*)carve((size_t)ND * HRD * 4);            // 32 MB
    unsigned short* Y   = (unsigned short*)carve((size_t)HRD * ND * 2);   // 16 MB
    float*          rno = (float*)carve((size_t)HRD * 4);

    pack_x_kernel<<<(ND * LRD) / 256, 256, 0, stream>>>(lr_x, Xb);
    pack_wt_kernel<<<(HRD * LRD) / 256, 256, 0, stream>>>(Wq, WtQ);
    pack_wt_kernel<<<(HRD * LRD) / 256, 256, 0, stream>>>(Wk, WtK);
    pack_wt_kernel<<<(HRD * LRD) / 256, 256, 0, stream>>>(Wv, WtV);
    pack_wt_kernel<<<(HRD * LRD) / 256, 256, 0, stream>>>(Wskip, WtS);

    dim3 pg(ND / 128, HRD / 64);
    proj_gemm_kernel<<<pg, 256, 0, stream>>>(Xb, WtQ, bq, (void*)Qb, 0);
    proj_gemm_kernel<<<pg, 256, 0, stream>>>(Xb, WtK, bk, (void*)Kb, 0);
    proj_gemm_kernel<<<pg, 256, 0, stream>>>(Xb, WtV, bv, (void*)VT, 1);
    proj_gemm_kernel<<<pg, 256, 0, stream>>>(Xb, WtS, bskip, (void*)OS, 2);

    attn_kernel<<<dim3(ND / 16, NHEAD), 256, 0, stream>>>(Qb, Kb, VT, OS);
    gn_kernel<<<HRD / 256, 256, 0, stream>>>(OS, gamma, beta, mscale, Y, rno);
    gram_kernel<<<dim3(HRD / 128, HRD / 64), 256, 0, stream>>>(Y, rno, G);
}
